// Blipv2_83047487635719
// MI455X (gfx1250) — matrix operand order repaired and checked by reading, not compiled, not run on hardware
//
#include <hip/hip_runtime.h>
#include <math.h>

#define BB   512
#define NN   65
#define DIM  768
#define AA   49
#define QKVD 2304
#define MLPD 3072
#define MTOT (BB*NN)                 // 33280
#define ATT_SCALE 0.03608439182435161f

typedef __bf16 bf16_t;
typedef __attribute__((ext_vector_type(16))) __bf16 v16bf;
typedef __attribute__((ext_vector_type(8)))  __bf16 v8bf;
typedef __attribute__((ext_vector_type(8)))  float  v8f;

// ---------------------------------------------------------------------------
// Pack f32 weight [Nout, K] (row-major, out = sum_k in[k]*W[n][k]) into the
// CDNA5 WMMA bf16 B-fragment layout: per (n16-tile, k32-tile): 32 lanes x 16
// bf16 contiguous.  b[lane][j] = W[n16*16 + lane%16][kt*32 + (j<8 ? (lane/16)*8 + j : 16 + (lane/16)*8 + j-8)]
// ---------------------------------------------------------------------------
__global__ void pack_weight_frags(const float* __restrict__ src,
                                  bf16_t* __restrict__ dst, int Nout, int K) {
  int gid = blockIdx.x * blockDim.x + threadIdx.x;
  int total = Nout * K;
  if (gid >= total) return;
  int j    = gid & 15;
  int lane = (gid >> 4) & 31;
  int frag = gid >> 9;
  int NTK  = K >> 5;
  int kt   = frag % NTK;
  int n16  = frag / NTK;
  int n = n16 * 16 + (lane & 15);
  int k = kt * 32 + (j < 8 ? (lane >> 4) * 8 + j : 16 + (lane >> 4) * 8 + (j - 8));
  dst[gid] = (bf16_t)src[(size_t)n * K + k];
}

// ---------------------------------------------------------------------------
// LayerNorm over DIM=768, one block per row, bf16 output
// ---------------------------------------------------------------------------
__launch_bounds__(256)
__global__ void layernorm_to_bf16(const float* __restrict__ x,
                                  const float* __restrict__ g,
                                  const float* __restrict__ bta,
                                  bf16_t* __restrict__ out) {
  __shared__ float red[256];
  int row = blockIdx.x;
  int t = threadIdx.x;
  const float* xr = x + (size_t)row * DIM;
  float v0[3];
  float s = 0.f;
#pragma unroll
  for (int i = 0; i < 3; i++) { float v = xr[t + 256 * i]; v0[i] = v; s += v; }
  red[t] = s; __syncthreads();
  for (int st = 128; st > 0; st >>= 1) { if (t < st) red[t] += red[t + st]; __syncthreads(); }
  float mean = red[0] / (float)DIM; __syncthreads();
  float s2 = 0.f;
#pragma unroll
  for (int i = 0; i < 3; i++) { float d = v0[i] - mean; s2 += d * d; }
  red[t] = s2; __syncthreads();
  for (int st = 128; st > 0; st >>= 1) { if (t < st) red[t] += red[t + st]; __syncthreads(); }
  float rstd = rsqrtf(red[0] / (float)DIM + 1e-5f);
#pragma unroll
  for (int i = 0; i < 3; i++) {
    int c = t + 256 * i;
    out[(size_t)row * DIM + c] = (bf16_t)((v0[i] - mean) * rstd * g[c] + bta[c]);
  }
}

// ---------------------------------------------------------------------------
// WMMA GEMM: C[M,N] = A[M,K] * W^T  with W pre-packed as B fragments.
// One wave computes a 32(M)x64(N) strip: 2 A fragments x 4 B tiles per K-step
// -> 8 independent v_wmma_f32_16x16x32_bf16 per iteration, B loaded once for
// two WMMAs.  Streamed A rows are prefetched ahead (global_prefetch_b8);
// packed weights are L2-resident (192 MB L2).
// epi: 0 = store bf16 (no bias)        [qkv]
//      1 = f32 store of C+bias+resid   [proj, fc2]
//      2 = bf16 store of gelu(C+bias)  [fc1]
// ---------------------------------------------------------------------------
__launch_bounds__(256)
__global__ void gemm_bf16_wmma(const bf16_t* __restrict__ A,
                               const bf16_t* __restrict__ Bf,
                               void* __restrict__ C,
                               const float* __restrict__ bias,
                               const float* __restrict__ resid,
                               int M, int N, int K, int epi) {
  int lane = threadIdx.x & 31;
  int gw = (int)((blockIdx.x * blockDim.x + threadIdx.x) >> 5);
  int ntn = N >> 6;
  int tm = gw / ntn;
  int tn = gw - tm * ntn;
  if (tm * 32 >= M) return;
  int NTK  = K >> 5;
  int half = lane >> 4;
  int l15  = lane & 15;

  const bf16_t* aBase0 = A + (size_t)(tm * 32 + l15) * K + half * 8;
  const bf16_t* aBase1 = aBase0 + (size_t)16 * K;
  const bf16_t* bBase[4];
#pragma unroll
  for (int s = 0; s < 4; s++)
    bBase[s] = Bf + ((size_t)(tn * 4 + s) * NTK) * 512 + lane * 16;

  v8f acc[8];
#pragma unroll
  for (int s = 0; s < 8; s++)
#pragma unroll
    for (int r = 0; r < 8; r++) acc[s][r] = 0.f;

  for (int kt = 0; kt < NTK; ++kt) {
    union { v16bf v; v8bf h[2]; } a0, a1;
    a0.h[0] = *(const v8bf*)(aBase0 + kt * 32);       // K = kt*32 + half*8 .. +7
    a0.h[1] = *(const v8bf*)(aBase0 + kt * 32 + 16);  // K = kt*32 + 16 + half*8 ..
    a1.h[0] = *(const v8bf*)(aBase1 + kt * 32);
    a1.h[1] = *(const v8bf*)(aBase1 + kt * 32 + 16);
    if (kt + 8 < NTK) {  // stream-ahead prefetch of the A rows
      __builtin_prefetch(aBase0 + (kt + 8) * 32, 0, 0);
      __builtin_prefetch(aBase1 + (kt + 8) * 32, 0, 0);
    }
#pragma unroll
    for (int s = 0; s < 4; s++) {
      v16bf b = *(const v16bf*)(bBase[s] + (size_t)kt * 512);
      acc[s] = __builtin_amdgcn_wmma_f32_16x16x32_bf16(
          false, a0.v, false, b, (short)0, acc[s], false, false);
      acc[4 + s] = __builtin_amdgcn_wmma_f32_16x16x32_bf16(
          false, a1.v, false, b, (short)0, acc[4 + s], false, false);
    }
  }

#pragma unroll
  for (int h = 0; h < 2; h++) {
    int rbase = tm * 32 + h * 16 + half * 8;
#pragma unroll
    for (int s = 0; s < 4; s++) {
      int col = tn * 64 + s * 16 + l15;
      float bv = bias ? bias[col] : 0.f;
#pragma unroll
      for (int r = 0; r < 8; r++) {
        size_t idx = (size_t)(rbase + r) * N + col;
        float v = acc[h * 4 + s][r] + bv;
        if (epi == 0) {
          ((bf16_t*)C)[idx] = (bf16_t)v;
        } else if (epi == 1) {
          ((float*)C)[idx] = v + resid[idx];
        } else {
          float gv = 0.5f * v * (1.f + erff(v * 0.70710678118654752f));
          ((bf16_t*)C)[idx] = (bf16_t)gv;
        }
      }
    }
  }
}

// ---------------------------------------------------------------------------
// Agent tokens: 2x2 average pooling of the 8x8 Q grid -> 7x7 agents (bf16)
// ---------------------------------------------------------------------------
__launch_bounds__(256)
__global__ void agent_pool(const bf16_t* __restrict__ qkv, bf16_t* __restrict__ agent) {
  int ba = blockIdx.x;
  int b = ba / AA, a = ba - b * AA;
  int i = a / 7, j = a - i * 7;
  size_t r00 = ((size_t)b * NN + 1 + i * 8 + j) * QKVD;
  size_t r01 = r00 + QKVD;
  size_t r10 = r00 + 8 * QKVD;
  size_t r11 = r10 + QKVD;
  for (int d = threadIdx.x; d < DIM; d += 256) {
    float v = 0.25f * ((float)qkv[r00 + d] + (float)qkv[r01 + d] +
                       (float)qkv[r10 + d] + (float)qkv[r11 + d]);
    agent[(size_t)ba * DIM + d] = (bf16_t)v;
  }
}

// ---------------------------------------------------------------------------
// Positional bias tables: pb[A][65] and ab[65][A] with bilinear 7->8 upsample
// ---------------------------------------------------------------------------
__device__ inline float bil78(const float* __restrict__ t, int y, int x) {
  float sy = fmaxf((y + 0.5f) * 0.875f - 0.5f, 0.f);
  float sx = fmaxf((x + 0.5f) * 0.875f - 0.5f, 0.f);
  int iy0 = (int)sy, ix0 = (int)sx;
  int iy1 = min(iy0 + 1, 6), ix1 = min(ix0 + 1, 6);
  float fy = sy - (float)iy0, fx = sx - (float)ix0;
  float v00 = t[iy0 * 7 + ix0], v01 = t[iy0 * 7 + ix1];
  float v10 = t[iy1 * 7 + ix0], v11 = t[iy1 * 7 + ix1];
  return (v00 * (1.f - fx) + v01 * fx) * (1.f - fy) +
         (v10 * (1.f - fx) + v11 * fx) * fy;
}

__global__ void build_biases(const float* __restrict__ an, const float* __restrict__ na,
                             const float* __restrict__ ah, const float* __restrict__ aw,
                             const float* __restrict__ ha, const float* __restrict__ wa,
                             const float* __restrict__ ac, const float* __restrict__ ca,
                             float* __restrict__ pb, float* __restrict__ ab) {
  int idx = blockIdx.x * blockDim.x + threadIdx.x;
  if (idx < AA * NN) {
    int a = idx / NN, n = idx - a * NN;
    float v;
    if (n == 0) v = ac[a];
    else {
      int np = n - 1, y = np >> 3, x = np & 7;
      v = bil78(an + a * 49, y, x) + ah[a * 8 + y] + aw[a * 8 + x];
    }
    pb[a * NN + n] = v;
  } else if (idx < 2 * AA * NN) {
    int i2 = idx - AA * NN;
    int n = i2 / AA, a = i2 - n * AA;
    float v;
    if (n == 0) v = ca[a];
    else {
      int np = n - 1, y = np >> 3, x = np & 7;
      v = bil78(na + a * 49, y, x) + ha[y * AA + a] + wa[x * AA + a];
    }
    ab[n * AA + a] = v;
  }
}

// ---------------------------------------------------------------------------
// agent_attn[b,a,:] = softmax_n( SCALE * agent.K^T + pb )  (written to d_out)
// ---------------------------------------------------------------------------
__launch_bounds__(128)
__global__ void agent_attn_kernel(const bf16_t* __restrict__ qkv,
                                  const bf16_t* __restrict__ agent,
                                  const float* __restrict__ pb,
                                  float* __restrict__ attn) {
  __shared__ float ag[DIM];
  __shared__ float red[128];
  int ba = blockIdx.x;
  int b = ba / AA, a = ba - b * AA;
  int t = threadIdx.x;
  for (int i = t; i < DIM; i += 128) ag[i] = (float)agent[(size_t)ba * DIM + i];
  __syncthreads();
  float logit = -1e30f;
  if (t < NN) {
    const bf16_t* kr = qkv + ((size_t)b * NN + t) * QKVD + DIM;
    float s = 0.f;
    for (int c = 0; c < DIM; c++) s += ag[c] * (float)kr[c];
    logit = s * ATT_SCALE + pb[a * NN + t];
  }
  red[t] = logit; __syncthreads();
  for (int st = 64; st > 0; st >>= 1) { if (t < st) red[t] = fmaxf(red[t], red[t + st]); __syncthreads(); }
  float mx = red[0]; __syncthreads();
  float e = (t < NN) ? __expf(logit - mx) : 0.f;
  red[t] = e; __syncthreads();
  for (int st = 64; st > 0; st >>= 1) { if (t < st) red[t] += red[t + st]; __syncthreads(); }
  float inv = 1.f / red[0];
  if (t < NN) attn[(size_t)ba * NN + t] = e * inv;
}

// ---------------------------------------------------------------------------
// agent_v[b,a,d] = sum_n attn[b,a,n] * V[b,n,d]
// ---------------------------------------------------------------------------
__launch_bounds__(256)
__global__ void agent_v_kernel(const float* __restrict__ attn,
                               const bf16_t* __restrict__ qkv,
                               float* __restrict__ agent_v) {
  __shared__ float at[NN];
  int ba = blockIdx.x;
  int b = ba / AA;
  int t = threadIdx.x;
  if (t < NN) at[t] = attn[(size_t)ba * NN + t];
  __syncthreads();
  for (int d = t; d < DIM; d += 256) {
    const bf16_t* vp = qkv + ((size_t)b * NN) * QKVD + 2 * DIM + d;
    float s = 0.f;
    for (int n = 0; n < NN; n++) s += at[n] * (float)vp[(size_t)n * QKVD];
    agent_v[(size_t)ba * DIM + d] = s;
  }
}

// ---------------------------------------------------------------------------
// q_attn[b,n,:] = softmax_a( SCALE * q.agent^T + ab )
// ---------------------------------------------------------------------------
__launch_bounds__(64)
__global__ void q_attn_kernel(const bf16_t* __restrict__ qkv,
                              const bf16_t* __restrict__ agent,
                              const float* __restrict__ ab,
                              float* __restrict__ qattn) {
  __shared__ float qr[DIM];
  __shared__ float red[64];
  int bn = blockIdx.x;
  int b = bn / NN, n = bn - b * NN;
  int t = threadIdx.x;
  for (int i = t; i < DIM; i += 64) qr[i] = (float)qkv[(size_t)bn * QKVD + i];
  __syncthreads();
  float logit = -1e30f;
  if (t < AA) {
    const bf16_t* ar = agent + ((size_t)b * AA + t) * DIM;
    float s = 0.f;
    for (int c = 0; c < DIM; c++) s += qr[c] * (float)ar[c];
    logit = s * ATT_SCALE + ab[n * AA + t];
  }
  red[t] = logit; __syncthreads();
  for (int st = 32; st > 0; st >>= 1) { if (t < st) red[t] = fmaxf(red[t], red[t + st]); __syncthreads(); }
  float mx = red[0]; __syncthreads();
  float e = (t < AA) ? __expf(logit - mx) : 0.f;
  red[t] = e; __syncthreads();
  for (int st = 32; st > 0; st >>= 1) { if (t < st) red[t] += red[t + st]; __syncthreads(); }
  float inv = 1.f / red[0];
  if (t < AA) qattn[(size_t)bn * AA + t] = e * inv;
}

// ---------------------------------------------------------------------------
// out[b,n,d] = sum_a qattn*agent_v  (+ depthwise 3x3 conv over V grid for n>0)
// stored bf16 as the A operand of the projection GEMM
// ---------------------------------------------------------------------------
__launch_bounds__(256)
__global__ void attn_out_kernel(const float* __restrict__ qattn,
                                const float* __restrict__ agent_v,
                                const bf16_t* __restrict__ qkv,
                                const float* __restrict__ dwc_w,
                                const float* __restrict__ dwc_b,
                                bf16_t* __restrict__ outb) {
  __shared__ float qa[AA];
  int bn = blockIdx.x;
  int b = bn / NN, n = bn - b * NN;
  int t = threadIdx.x;
  if (t < AA) qa[t] = qattn[(size_t)bn * AA + t];
  __syncthreads();
  int y = (n - 1) >> 3, x = (n - 1) & 7;
  for (int d = t; d < DIM; d += 256) {
    const float* av = agent_v + (size_t)b * AA * DIM + d;
    float s = 0.f;
    for (int a = 0; a < AA; a++) s += qa[a] * av[(size_t)a * DIM];
    if (n > 0) {
      float cs = dwc_b[d];
#pragma unroll
      for (int dy = -1; dy <= 1; dy++) {
        int yy = y + dy; if (yy < 0 || yy > 7) continue;
#pragma unroll
        for (int dx = -1; dx <= 1; dx++) {
          int xx = x + dx; if (xx < 0 || xx > 7) continue;
          float vv = (float)qkv[((size_t)b * NN + 1 + yy * 8 + xx) * QKVD + 2 * DIM + d];
          cs += vv * dwc_w[d * 9 + (dy + 1) * 3 + (dx + 1)];
        }
      }
      s += cs;
    }
    outb[(size_t)bn * DIM + d] = (bf16_t)s;
  }
}

// ---------------------------------------------------------------------------
extern "C" void kernel_launch(void* const* d_in, const int* in_sizes, int n_in,
                              void* d_out, int out_size, void* d_ws, size_t ws_size,
                              hipStream_t stream) {
  (void)in_sizes; (void)n_in; (void)out_size; (void)ws_size;
  const float* x      = (const float*)d_in[0];
  const float* w_qkv  = (const float*)d_in[1];
  const float* w_proj = (const float*)d_in[2];
  const float* b_proj = (const float*)d_in[3];
  const float* dwc_w  = (const float*)d_in[4];
  const float* dwc_b  = (const float*)d_in[5];
  const float* an     = (const float*)d_in[6];
  const float* na     = (const float*)d_in[7];
  const float* ah     = (const float*)d_in[8];
  const float* aw     = (const float*)d_in[9];
  const float* ha     = (const float*)d_in[10];
  const float* wa     = (const float*)d_in[11];
  const float* ac     = (const float*)d_in[12];
  const float* ca     = (const float*)d_in[13];
  const float* ln1g   = (const float*)d_in[14];
  const float* ln1b   = (const float*)d_in[15];
  const float* ln2g   = (const float*)d_in[16];
  const float* ln2b   = (const float*)d_in[17];
  const float* fc1w   = (const float*)d_in[18];
  const float* fc1b   = (const float*)d_in[19];
  const float* fc2w   = (const float*)d_in[20];
  const float* fc2b   = (const float*)d_in[21];

  float* x_out      = (float*)d_out;                       // [B,N,DIM]
  float* agent_rep  = (float*)d_out + (size_t)MTOT * DIM;  // [B,1,A,N]

  char* wp = (char*)d_ws;
  size_t off = 0;
  auto alloc = [&](size_t bytes) -> char* {
    char* p = wp + off;
    off += (bytes + 255) & ~(size_t)255;
    return p;
  };
  bf16_t* wqkv_f  = (bf16_t*)alloc((size_t)QKVD * DIM * 2);
  bf16_t* wproj_f = (bf16_t*)alloc((size_t)DIM * DIM * 2);
  bf16_t* wfc1_f  = (bf16_t*)alloc((size_t)MLPD * DIM * 2);
  bf16_t* wfc2_f  = (bf16_t*)alloc((size_t)DIM * MLPD * 2);
  bf16_t* xn      = (bf16_t*)alloc((size_t)MTOT * DIM * 2);
  bf16_t* qkv     = (bf16_t*)alloc((size_t)MTOT * QKVD * 2);
  bf16_t* agent   = (bf16_t*)alloc((size_t)BB * AA * DIM * 2);
  float*  pb      = (float*) alloc((size_t)AA * NN * 4);
  float*  ab      = (float*) alloc((size_t)NN * AA * 4);
  float*  agentv  = (float*) alloc((size_t)BB * AA * DIM * 4);
  float*  qattn   = (float*) alloc((size_t)MTOT * AA * 4);
  bf16_t* out_att = (bf16_t*)alloc((size_t)MTOT * DIM * 2);
  float*  x1      = (float*) alloc((size_t)MTOT * DIM * 4);
  bf16_t* h2      = (bf16_t*)alloc((size_t)MTOT * DIM * 2);
  bf16_t* hmlp    = (bf16_t*)alloc((size_t)MTOT * MLPD * 2);

  // 1. pack weights into WMMA B-fragment layout
  pack_weight_frags<<<(QKVD * DIM + 255) / 256, 256, 0, stream>>>(w_qkv, wqkv_f, QKVD, DIM);
  pack_weight_frags<<<(DIM * DIM + 255) / 256, 256, 0, stream>>>(w_proj, wproj_f, DIM, DIM);
  pack_weight_frags<<<(MLPD * DIM + 255) / 256, 256, 0, stream>>>(fc1w, wfc1_f, MLPD, DIM);
  pack_weight_frags<<<(DIM * MLPD + 255) / 256, 256, 0, stream>>>(fc2w, wfc2_f, DIM, MLPD);

  // 2. LN1 -> bf16
  layernorm_to_bf16<<<MTOT, 256, 0, stream>>>(x, ln1g, ln1b, xn);

  // 3. QKV GEMM: [33280x768] x [768x2304] -> bf16   (32x64 per wave)
  gemm_bf16_wmma<<<(MTOT / 32) * (QKVD / 64) / 8, 256, 0, stream>>>(
      xn, wqkv_f, qkv, nullptr, nullptr, MTOT, QKVD, DIM, 0);

  // 4. agents + bias tables
  agent_pool<<<BB * AA, 256, 0, stream>>>(qkv, agent);
  build_biases<<<(2 * AA * NN + 255) / 256, 256, 0, stream>>>(an, na, ah, aw, ha, wa, ac, ca, pb, ab);

  // 5. attention
  agent_attn_kernel<<<BB * AA, 128, 0, stream>>>(qkv, agent, pb, agent_rep);
  agent_v_kernel<<<BB * AA, 256, 0, stream>>>(agent_rep, qkv, agentv);
  q_attn_kernel<<<MTOT, 64, 0, stream>>>(qkv, agent, ab, qattn);
  attn_out_kernel<<<MTOT, 256, 0, stream>>>(qattn, agentv, qkv, dwc_w, dwc_b, out_att);

  // 6. proj GEMM + bias + residual -> x1 (f32)
  gemm_bf16_wmma<<<(MTOT / 32) * (DIM / 64) / 8, 256, 0, stream>>>(
      out_att, wproj_f, x1, b_proj, x, MTOT, DIM, DIM, 1);

  // 7. LN2 -> bf16
  layernorm_to_bf16<<<MTOT, 256, 0, stream>>>(x1, ln2g, ln2b, h2);

  // 8. fc1 GEMM + bias + exact GELU -> bf16
  gemm_bf16_wmma<<<(MTOT / 32) * (MLPD / 64) / 8, 256, 0, stream>>>(
      h2, wfc1_f, hmlp, fc1b, nullptr, MTOT, MLPD, DIM, 2);

  // 9. fc2 GEMM + bias + residual(x1) -> final output (f32)
  gemm_bf16_wmma<<<(MTOT / 32) * (DIM / 64) / 8, 256, 0, stream>>>(
      hmlp, wfc2_f, x_out, fc2b, x1, MTOT, DIM, MLPD, 1);
}
